// Dinov2SdpaSelfAttention_Wrapper_30794915512806
// MI455X (gfx1250) — matrix operand order repaired and checked by reading, not compiled, not run on hardware
//
#include <hip/hip_runtime.h>
#include <hip/hip_bf16.h>

typedef __bf16 bf16_t;
typedef __attribute__((ext_vector_type(16))) __bf16 v16bf;
typedef __attribute__((ext_vector_type(8)))  __bf16 v8bf;
typedef __attribute__((ext_vector_type(8)))  float  v8f;
typedef __attribute__((ext_vector_type(4)))  float  v4f;

#define BATCH 4
#define LSEQ  1370
#define HID   1024
#define NH    16
#define HD    64
#define MTOT  (BATCH * LSEQ)   // 5480
#define SPAD  1376             // LSEQ padded to multiple of 32

__device__ __forceinline__ v8f wmma_bf16(v16bf a, v16bf b, v8f c) {
  return __builtin_amdgcn_wmma_f32_16x16x32_bf16(false, a, false, b, (short)0, c,
                                                 false, false);
}

// A-fragment from bf16 row: two contiguous 8-element chunks (p[0..7], p[8..15])
__device__ __forceinline__ v16bf load_a_bf(const bf16_t* p) {
  v8bf lo = *(const v8bf*)p;
  v8bf hi = *(const v8bf*)(p + 8);
  return __builtin_shufflevector(lo, hi, 0, 1, 2, 3, 4, 5, 6, 7, 8, 9, 10, 11,
                                 12, 13, 14, 15);
}
// B-fragment from bf16 row: 16 contiguous values (32B aligned)
__device__ __forceinline__ v16bf load_b_bf(const bf16_t* p) {
  return *(const v16bf*)p;
}

// ---------------------------------------------------------------------------
// Phase 0: one-shot fp32 -> bf16 conversion (X and the three weight matrices).
// Memory bound; done once per call so the GEMM inner loop is cvt-free.
// ---------------------------------------------------------------------------
__global__ __launch_bounds__(256) void cvt_bf16_kernel(
    const float* __restrict__ src, bf16_t* __restrict__ dst, int n8) {
  const int i = blockIdx.x * blockDim.x + threadIdx.x;
  if (i >= n8) return;
  const float* p = src + (size_t)i * 8;
  v4f a = *(const v4f*)p;
  v4f b = *(const v4f*)(p + 4);
  v8bf r;
#pragma unroll
  for (int e = 0; e < 4; ++e) {
    r[e]     = (bf16_t)a[e];
    r[4 + e] = (bf16_t)b[e];
  }
  *(v8bf*)(dst + (size_t)i * 8) = r;
}

// ---------------------------------------------------------------------------
// Phase 1: fused QKV projection.  y = X * W^T + b  (torch Linear), bf16 in.
// Each wave: 32(M) x 64(N) tile, computing Q,K,V together.
//   - X A-fragments reused 12x each, W B-fragments reused 2x each.
//   - 24 f32 accumulators (192 VGPRs); 4 waves per block.
// Outputs bf16: Q[bh][l][64], K[bh][l][64], Vt[bh][d][SPAD].
// ---------------------------------------------------------------------------
__global__ __launch_bounds__(128) void qkv_proj_kernel(
    const bf16_t* __restrict__ X,
    const bf16_t* __restrict__ Wq, const float* __restrict__ bq,
    const bf16_t* __restrict__ Wk, const float* __restrict__ bk,
    const bf16_t* __restrict__ Wv, const float* __restrict__ bv,
    bf16_t* __restrict__ Qo, bf16_t* __restrict__ Ko, bf16_t* __restrict__ Vt) {
  const int lane  = threadIdx.x & 31;
  const int wid   = (blockIdx.x * blockDim.x + threadIdx.x) >> 5;
  const int mtile = wid >> 4;   // 0..171 (32 rows each)
  const int ncol  = wid & 15;   // 0..15 -> 64 columns each
  const int m0 = mtile * 32;
  const int n0 = ncol * 64;
  const int lh = lane & 15;
  const int hi = lane >> 4;
  const int abase = hi ? 16 : 0;
  const int bbase = hi ? 16 : 0;

  int arow[2];
#pragma unroll
  for (int s = 0; s < 2; ++s) arow[s] = min(m0 + s * 16 + lh, MTOT - 1);

  v8f accQ[2][4], accK[2][4], accV[2][4];
#pragma unroll
  for (int nt = 0; nt < 4; ++nt) {
    const int n = n0 + nt * 16 + lh;
    const float q = bq[n], k = bk[n], v = bv[n];
#pragma unroll
    for (int s = 0; s < 2; ++s)
#pragma unroll
      for (int e = 0; e < 8; ++e) {
        accQ[s][nt][e] = q; accK[s][nt][e] = k; accV[s][nt][e] = v;
      }
  }

  for (int k0 = 0; k0 < HID; k0 += 32) {
    v16bf af[2];
#pragma unroll
    for (int s = 0; s < 2; ++s)
      af[s] = load_a_bf(X + (size_t)arow[s] * HID + k0 + abase);
#pragma unroll
    for (int nt = 0; nt < 4; ++nt) {
      const int n = n0 + nt * 16 + lh;
      const size_t off = (size_t)n * HID + k0 + bbase;
      const v16bf bqf = load_b_bf(Wq + off);
      accQ[0][nt] = wmma_bf16(af[0], bqf, accQ[0][nt]);
      accQ[1][nt] = wmma_bf16(af[1], bqf, accQ[1][nt]);
      const v16bf bkf = load_b_bf(Wk + off);
      accK[0][nt] = wmma_bf16(af[0], bkf, accK[0][nt]);
      accK[1][nt] = wmma_bf16(af[1], bkf, accK[1][nt]);
      const v16bf bvf = load_b_bf(Wv + off);
      accV[0][nt] = wmma_bf16(af[0], bvf, accV[0][nt]);
      accV[1][nt] = wmma_bf16(af[1], bvf, accV[1][nt]);
    }
  }

#pragma unroll
  for (int nt = 0; nt < 4; ++nt) {
    const int n = n0 + nt * 16 + lh;
    const int h = n >> 6, d = n & 63;
#pragma unroll
    for (int s = 0; s < 2; ++s)
#pragma unroll
      for (int r = 0; r < 8; ++r) {
        const int m = m0 + s * 16 + (hi ? 8 : 0) + r;
        if (m < MTOT) {
          const int b  = m / LSEQ;
          const int l  = m - b * LSEQ;
          const int bh = b * NH + h;
          Qo[((size_t)bh * LSEQ + l) * HD + d] = (bf16_t)accQ[s][nt][r];
          Ko[((size_t)bh * LSEQ + l) * HD + d] = (bf16_t)accK[s][nt][r];
          Vt[((size_t)bh * HD + d) * SPAD + l] = (bf16_t)accV[s][nt][r];
        }
      }
  }
}

// ---------------------------------------------------------------------------
// Phase 2: flash-style attention. One wave per 16 query rows per (b,h).
// Streams S in tiles of 32; online softmax; P transposed C->A layout via LDS.
// ---------------------------------------------------------------------------
__global__ __launch_bounds__(256) void attn_kernel(
    const bf16_t* __restrict__ Q, const bf16_t* __restrict__ K,
    const bf16_t* __restrict__ Vt, float* __restrict__ out) {
  __shared__ __align__(32) bf16_t pshare[8][16 * 32];

  const int lane  = threadIdx.x & 31;
  const int wslot = threadIdx.x >> 5;
  const int bh = blockIdx.x / 11;
  const int mt = (blockIdx.x % 11) * 8 + wslot;   // 0..87 (86 valid)
  const int m0 = mt * 16;
  if (m0 >= LSEQ) return;
  const int b = bh >> 4, h = bh & 15;
  const int lh = lane & 15;
  const int hi = lane >> 4;
  const int abase = hi ? 16 : 0;
  const int bbase = hi ? 16 : 0;

  const bf16_t* Qb = Q + (size_t)bh * LSEQ * HD;
  const bf16_t* Kb = K + (size_t)bh * LSEQ * HD;
  const bf16_t* Vb = Vt + (size_t)bh * HD * SPAD;

  // Q A-fragments for K'={0..31},{32..63}: fixed over the whole S loop.
  const int qrow = min(m0 + lh, LSEQ - 1);
  v16bf qa[2];
#pragma unroll
  for (int t = 0; t < 2; ++t)
    qa[t] = load_a_bf(Qb + (size_t)qrow * HD + t * 32 + abase);

  v8f acc[4];
  float rmax[8], rsum[8];
#pragma unroll
  for (int j = 0; j < 4; ++j)
#pragma unroll
    for (int e = 0; e < 8; ++e) acc[j][e] = 0.f;
#pragma unroll
  for (int r = 0; r < 8; ++r) { rmax[r] = -1e30f; rsum[r] = 0.f; }

  bf16_t* pw = &pshare[wslot][0];
  const float scale = 0.125f;   // 1/sqrt(64)

  for (int s0 = 0; s0 < SPAD; s0 += 32) {
    // ---- scores: two 16x16 C tiles covering S columns s0..s0+31 ----
    v8f c0, c1;
#pragma unroll
    for (int e = 0; e < 8; ++e) { c0[e] = 0.f; c1[e] = 0.f; }
    const int sr0 = min(s0 + lh, LSEQ - 1);
    const int sr1 = min(s0 + 16 + lh, LSEQ - 1);
#pragma unroll
    for (int t = 0; t < 2; ++t) {
      const int kk = t * 32;
      c0 = wmma_bf16(qa[t], load_b_bf(Kb + (size_t)sr0 * HD + kk + bbase), c0);
      c1 = wmma_bf16(qa[t], load_b_bf(Kb + (size_t)sr1 * HD + kk + bbase), c1);
    }

    // ---- online softmax (rows live across 16-lane groups) ----
    const bool v0 = (s0 + lh) < LSEQ;
    const bool v1 = (s0 + 16 + lh) < LSEQ;
    float p0[8], p1[8];
#pragma unroll
    for (int r = 0; r < 8; ++r) {
      float x0 = v0 ? c0[r] * scale : -1e30f;
      float x1 = v1 ? c1[r] * scale : -1e30f;
      float mx = fmaxf(x0, x1);
#pragma unroll
      for (int off = 1; off < 16; off <<= 1)
        mx = fmaxf(mx, __shfl_xor(mx, off, 32));
      const float mnew  = fmaxf(rmax[r], mx);
      const float alpha = __expf(rmax[r] - mnew);
      rmax[r] = mnew;
      const float e0 = __expf(x0 - mnew);
      const float e1 = __expf(x1 - mnew);
      float ssum = e0 + e1;
#pragma unroll
      for (int off = 1; off < 16; off <<= 1)
        ssum += __shfl_xor(ssum, off, 32);
      rsum[r] = rsum[r] * alpha + ssum;
      p0[r] = e0;
      p1[r] = e1;
#pragma unroll
      for (int j = 0; j < 4; ++j) acc[j][r] *= alpha;
    }

    // ---- transpose P (C layout -> A layout) through wave-private LDS ----
#pragma unroll
    for (int r = 0; r < 8; ++r) {
      const int mr = (hi ? 8 : 0) + r;
      pw[mr * 32 + lh]      = (bf16_t)p0[r];
      pw[mr * 32 + lh + 16] = (bf16_t)p1[r];
    }
    asm volatile("s_wait_dscnt 0" ::: "memory");
    const v16bf pa = load_a_bf(pw + lh * 32 + abase);

    // ---- ctx += P * V  (B = V, Bt = Vt rows, contiguous in S) ----
#pragma unroll
    for (int j = 0; j < 4; ++j) {
      const bf16_t* vp = Vb + (size_t)(j * 16 + lh) * SPAD + s0 + bbase;
      acc[j] = wmma_bf16(pa, load_b_bf(vp), acc[j]);
    }
  }

  // ---- normalize + store fp32 out[b][l][h*64+d] ----
#pragma unroll
  for (int j = 0; j < 4; ++j) {
    const int d = j * 16 + lh;
#pragma unroll
    for (int r = 0; r < 8; ++r) {
      const int lq = m0 + (hi ? 8 : 0) + r;
      if (lq < LSEQ)
        out[((size_t)b * LSEQ + lq) * HID + h * HD + d] = acc[j][r] / rsum[r];
    }
  }
}

extern "C" void kernel_launch(void* const* d_in, const int* in_sizes, int n_in,
                              void* d_out, int out_size, void* d_ws, size_t ws_size,
                              hipStream_t stream) {
  const float* X  = (const float*)d_in[0];
  const float* Wq = (const float*)d_in[1];
  const float* bq = (const float*)d_in[2];
  const float* Wk = (const float*)d_in[3];
  const float* bk = (const float*)d_in[4];
  const float* Wv = (const float*)d_in[5];
  const float* bv = (const float*)d_in[6];
  float* out = (float*)d_out;

  const size_t nQK = (size_t)BATCH * NH * LSEQ * HD;  // 5,611,520 bf16 elements
  const size_t nVt = (size_t)BATCH * NH * HD * SPAD;  // 5,636,096
  const size_t nX  = (size_t)MTOT * HID;              // 5,611,520
  const size_t nW  = (size_t)HID * HID;               // 1,048,576

  bf16_t* Qo  = (bf16_t*)d_ws;
  bf16_t* Ko  = Qo + nQK;
  bf16_t* Vt  = Ko + nQK;
  bf16_t* Xb  = Vt + nVt;
  bf16_t* Wqb = Xb + nX;
  bf16_t* Wkb = Wqb + nW;
  bf16_t* Wvb = Wkb + nW;

  // Zero Vt so the SPAD padding columns contribute exact zeros (capture-safe).
  hipMemsetAsync(Vt, 0, nVt * sizeof(bf16_t), stream);

  // Phase 0: fp32 -> bf16 one-shot conversions (8 elements/thread).
  {
    const int tx = 256;
    const int nx8 = (int)(nX / 8), nw8 = (int)(nW / 8);
    cvt_bf16_kernel<<<(nx8 + tx - 1) / tx, tx, 0, stream>>>(X, Xb, nx8);
    cvt_bf16_kernel<<<(nw8 + tx - 1) / tx, tx, 0, stream>>>(Wq, Wqb, nw8);
    cvt_bf16_kernel<<<(nw8 + tx - 1) / tx, tx, 0, stream>>>(Wk, Wkb, nw8);
    cvt_bf16_kernel<<<(nw8 + tx - 1) / tx, tx, 0, stream>>>(Wv, Wvb, nw8);
  }

  // Phase 1: 172 M-tiles (32 rows) x 16 N-columns = 2752 waves, 4 waves/block.
  qkv_proj_kernel<<<688, 128, 0, stream>>>(Xb, Wqb, bq, Wkb, bk, Wvb, bv,
                                           Qo, Ko, Vt);

  // Phase 2: 64 (b,h) x 11 blocks (8 waves each -> 88 m-tiles, 86 valid).
  attn_kernel<<<64 * 11, 256, 0, stream>>>(Qo, Ko, Vt, out);
}
